// DGMC_36395552866760
// MI455X (gfx1250) — matrix operand order repaired and checked by reading, not compiled, not run on hardware
//
#include <hip/hip_runtime.h>
#include <math.h>

// ---------------- problem constants (match reference) ----------------
constexpr int NS    = 20000;
constexpr int NT    = 20000;
constexpr int EDG   = 320000;
constexpr int C_IN  = 128;
constexpr int C_OUT = 256;
constexpr int RDIM  = 32;   // R_IN == R_OUT == 32
constexpr int TOPK  = 10;

// ---------------- CDNA5 WMMA vector types ----------------
typedef __attribute__((ext_vector_type(16))) __bf16 v16bf;
typedef __attribute__((ext_vector_type(16))) short  v16s;
typedef __attribute__((ext_vector_type(8)))  short  v8s;
typedef __attribute__((ext_vector_type(8)))  float  v8f;

// f32 -> bf16 round-to-nearest-even (bit pattern)
static __device__ __forceinline__ short f2bf(float f) {
    unsigned u = __builtin_bit_cast(unsigned, f);
    unsigned r = u + 0x7fffu + ((u >> 16) & 1u);
    return (short)(r >> 16);
}

// A-matrix tile loader (16-bit A 16x32, contiguous K order matching load_b16, not the ISA table order):
// lanes 0-15 hold K=0..15, lanes 16-31 hold K=16..31
static __device__ __forceinline__ v16bf load_a16(const unsigned short* rowptr, int kt, int laneHi) {
    int base = kt * 32 + (laneHi ? 16 : 0);
    v8s lo = *(const v8s*)(rowptr + base);
    v8s hi = *(const v8s*)(rowptr + base + 8);
    v16s s;
#pragma unroll
    for (int i = 0; i < 8; ++i) { s[i] = lo[i]; s[i + 8] = hi[i]; }
    return __builtin_bit_cast(v16bf, s);
}

// B-matrix tile loader (K-contiguous per lane-half: lanes 0-15 K=0..15, lanes 16-31 K=16..31)
static __device__ __forceinline__ v16bf load_b16(const unsigned short* rowptr, int kt, int laneHi) {
    const v16s* p = (const v16s*)(rowptr + kt * 32 + (laneHi ? 16 : 0));
    return __builtin_bit_cast(v16bf, *p);
}

// ---------------- threefry2x32 (exact JAX PRNG core) ----------------
__host__ __device__ inline unsigned rotl32(unsigned x, int r) { return (x << r) | (x >> (32 - r)); }
__host__ __device__ inline void threefry2x32(unsigned k0, unsigned k1, unsigned c0, unsigned c1,
                                             unsigned& o0, unsigned& o1) {
    unsigned ks[3] = { k0, k1, 0x1BD11BDAu ^ k0 ^ k1 };
    unsigned x0 = c0 + k0, x1 = c1 + k1;
    const int R[8] = { 13, 15, 26, 6, 17, 29, 16, 24 };
    for (int i = 0; i < 20; ++i) {
        x0 += x1; x1 = rotl32(x1, R[i & 7]); x1 ^= x0;
        if ((i & 3) == 3) {
            int j = i / 4 + 1;
            x0 += ks[j % 3];
            x1 += ks[(j + 1) % 3] + (unsigned)j;
        }
    }
    o0 = x0; o1 = x1;
}

// bits -> N(0,1): uniform(-1,1) then sqrt(2)*erfinv (Giles single-precision poly)
static __device__ __forceinline__ float bits_to_normal(unsigned bits) {
    float f = __builtin_bit_cast(float, 0x3f800000u | (bits >> 9)) - 1.0f; // [0,1)
    const float lo = -0.99999994f;
    float u = f * (1.0f - lo) + lo;
    float w = -__logf((1.0f - u) * (1.0f + u));
    float p;
    if (w < 5.0f) {
        w -= 2.5f;
        p = 2.81022636e-08f;
        p = fmaf(p, w, 3.43273939e-07f);
        p = fmaf(p, w, -3.5233877e-06f);
        p = fmaf(p, w, -4.39150654e-06f);
        p = fmaf(p, w, 0.00021858087f);
        p = fmaf(p, w, -0.00125372503f);
        p = fmaf(p, w, -0.00417768164f);
        p = fmaf(p, w, 0.246640727f);
        p = fmaf(p, w, 1.50140941f);
    } else {
        w = sqrtf(w) - 3.0f;
        p = -0.000200214257f;
        p = fmaf(p, w, 0.000100950558f);
        p = fmaf(p, w, 0.00134934322f);
        p = fmaf(p, w, -0.00367342844f);
        p = fmaf(p, w, 0.00573950773f);
        p = fmaf(p, w, -0.0076224613f);
        p = fmaf(p, w, 0.00943887047f);
        p = fmaf(p, w, 1.00167406f);
        p = fmaf(p, w, 2.83297682f);
    }
    return 1.41421356f * p * u;
}

// ================= kernels =================

// edge scatter: msg[col] += x[row] * w_e, 128 channels (4 per thread)
__global__ void k_scatter128(const float* __restrict__ x, const int* __restrict__ ei,
                             const float* __restrict__ ea, float* __restrict__ msg) {
    int t = blockIdx.x * blockDim.x + threadIdx.x;
    int e = t >> 5, c4 = t & 31;
    if (e >= EDG) return;
    int row = ei[e];
    int col = ei[EDG + e];
    float w = ea[e];
    const float4 xv = *(const float4*)(x + row * C_IN + c4 * 4);
    float* m = msg + col * C_IN + c4 * 4;
    atomicAdd(m + 0, xv.x * w);
    atomicAdd(m + 1, xv.y * w);
    atomicAdd(m + 2, xv.z * w);
    atomicAdd(m + 3, xv.w * w);
}

// pack psi1_W (f32 [128][256]) into WMMA-B bf16 layout: [kt][nt][lane][elem]
__global__ void k_packW(const float* __restrict__ W, unsigned short* __restrict__ Wp) {
    int idx = blockIdx.x * blockDim.x + threadIdx.x;
    if (idx >= 4 * 16 * 32 * 16) return;
    int e = idx & 15;
    int lane = (idx >> 4) & 31;
    int t = idx >> 9;            // kt*16 + nt
    int nt = t & 15, kt = t >> 4;
    int n = nt * 16 + (lane & 15);
    int k = kt * 32 + ((lane >= 16) ? 8 : 0) + (e < 8 ? e : 16 + (e - 8));
    Wp[idx] = (unsigned short)f2bf(W[k * C_OUT + n]);
}

// h = relu((x+msg) @ W + b), via WMMA bf16; also emits bf16 copy of h
__global__ __launch_bounds__(128) void k_gnn1_gemm(const float* __restrict__ x,
                                                   const float* __restrict__ msg,
                                                   const unsigned short* __restrict__ Wp,
                                                   const float* __restrict__ b,
                                                   float* __restrict__ h,
                                                   unsigned short* __restrict__ h_bf) {
    int tid = threadIdx.x, wave = tid >> 5, lane = tid & 31;
    int laneHi = (lane >= 16), lmod = lane & 15;
    int m0 = blockIdx.x * 16;
    int nt = blockIdx.y * 4 + wave;       // n-tile 0..15
    int n0 = nt * 16;
    const float* xr = x + (m0 + lmod) * C_IN;
    const float* mr = msg + (m0 + lmod) * C_IN;

    float bias = b[n0 + lmod];
    v8f c;
#pragma unroll
    for (int j = 0; j < 8; ++j) c[j] = bias;

#pragma unroll
    for (int kt = 0; kt < 4; ++kt) {
        int kb = kt * 32 + (laneHi ? 8 : 0);
        v16s s;
#pragma unroll
        for (int i = 0; i < 8; ++i) {
            s[i]     = f2bf(xr[kb + i]      + mr[kb + i]);
            s[i + 8] = f2bf(xr[kb + 16 + i] + mr[kb + 16 + i]);
        }
        v16bf a = __builtin_bit_cast(v16bf, s);
        const v16s* bp = (const v16s*)(Wp + (((size_t)kt * 16 + nt) * 32 + lane) * 16);
        v16bf bm = __builtin_bit_cast(v16bf, *bp);
        c = __builtin_amdgcn_wmma_f32_16x16x32_bf16(false, a, false, bm, (short)0, c, false, false);
    }
#pragma unroll
    for (int j = 0; j < 8; ++j) {
        int rr = m0 + j + (laneHi ? 8 : 0);
        int cc = n0 + lmod;
        float v = fmaxf(c[j], 0.0f);
        h[rr * C_OUT + cc] = v;
        h_bf[rr * C_OUT + cc] = (unsigned short)f2bf(v);
    }
}

// WMMA similarity + fused streaming top-10 (the 205-GFLOP hot loop)
__global__ __launch_bounds__(128) void k_sim_topk(const unsigned short* __restrict__ hs_bf,
                                                  const unsigned short* __restrict__ ht_bf,
                                                  int* __restrict__ S_idx) {
    __shared__ float lds_sim[16 * 64];
    __shared__ float lds_v[16 * 8 * TOPK];
    __shared__ int   lds_i[16 * 8 * TOPK];

    int tid = threadIdx.x;
    int wave = tid >> 5, lane = tid & 31;
    int laneHi = (lane >= 16), lmod = lane & 15;
    int m0 = blockIdx.x * 16;

    // hoist A tiles (16 s-rows, full K=256) into registers for the whole sweep
    const unsigned short* arow = hs_bf + (size_t)(m0 + lmod) * C_OUT;
    v16bf a[8];
#pragma unroll
    for (int kt = 0; kt < 8; ++kt) a[kt] = load_a16(arow, kt, laneHi);

    float tv[TOPK]; int ti[TOPK];
#pragma unroll
    for (int j = 0; j < TOPK; ++j) { tv[j] = -1e30f; ti[j] = 0; }
    float curmin = -1e30f;

    int row = tid >> 3;   // 0..15: s-row within tile
    int seg = tid & 7;    // 0..7 : 8-column segment

    for (int t0 = 0; t0 < NT; t0 += 64) {
        int tcol = t0 + wave * 16 + lmod;
        int tc = tcol < NT ? tcol : NT - 1;               // clamp, keep EXEC all-1 for WMMA
        const unsigned short* brow = ht_bf + (size_t)tc * C_OUT;
        v8f c;
#pragma unroll
        for (int j = 0; j < 8; ++j) c[j] = 0.0f;
#pragma unroll
        for (int kt = 0; kt < 8; ++kt) {
            v16bf bm = load_b16(brow, kt, laneHi);
            c = __builtin_amdgcn_wmma_f32_16x16x32_bf16(false, a[kt], false, bm, (short)0, c, false, false);
        }
#pragma unroll
        for (int j = 0; j < 8; ++j)
            lds_sim[(j + (laneHi ? 8 : 0)) * 64 + wave * 16 + lmod] = c[j];
        __syncthreads();

#pragma unroll
        for (int cc = 0; cc < 8; ++cc) {
            int col = seg * 8 + cc;
            float v = lds_sim[row * 64 + col];
            int gidx = t0 + col;
            if (gidx < NT && v > curmin) {
                int mj = 0; float mv = tv[0];
#pragma unroll
                for (int j = 1; j < TOPK; ++j) if (tv[j] < mv) { mv = tv[j]; mj = j; }
                tv[mj] = v; ti[mj] = gidx;
                curmin = tv[0];
#pragma unroll
                for (int j = 1; j < TOPK; ++j) curmin = fminf(curmin, tv[j]);
            }
        }
        __syncthreads();
    }

    // dump 8 partial lists per row, then select top-10 (desc, index tie-break)
#pragma unroll
    for (int j = 0; j < TOPK; ++j) {
        lds_v[(row * 8 + seg) * TOPK + j] = tv[j];
        lds_i[(row * 8 + seg) * TOPK + j] = ti[j];
    }
    __syncthreads();
    if (tid < 16) {
        int r = tid;
        for (int j = 0; j < TOPK; ++j) {
            float bv = -1e38f; int bi = 0x7fffffff; int bp = 0;
            for (int p = 0; p < 8 * TOPK; ++p) {
                float v = lds_v[r * 80 + p];
                int  id = lds_i[r * 80 + p];
                if (v > bv || (v == bv && id < bi)) { bv = v; bi = id; bp = p; }
            }
            S_idx[(m0 + r) * TOPK + j] = bi;
            lds_v[r * 80 + bp] = -1e38f;
        }
    }
}

// exact f32 S_hat[n][j] = <h_s[n], h_t[S_idx[n][j]]>, one wave per n
__global__ __launch_bounds__(256) void k_shat(const float* __restrict__ hs, const float* __restrict__ ht,
                                              const int* __restrict__ S_idx, float* __restrict__ S_hat) {
    int tid = threadIdx.x, lane = tid & 31, w = tid >> 5;
    int n = blockIdx.x * 8 + w;
    if (n >= NS) return;
    const float* a = hs + (size_t)n * C_OUT;
#pragma unroll
    for (int j = 0; j < TOPK; ++j) {
        int t = S_idx[n * TOPK + j];
        const float* bptr = ht + (size_t)t * C_OUT;
        float acc = 0.0f;
#pragma unroll
        for (int cc = 0; cc < 8; ++cc) acc += a[lane + cc * 32] * bptr[lane + cc * 32];
        for (int off = 16; off; off >>= 1) acc += __shfl_xor(acc, off, 32);
        if (lane == 0) S_hat[n * TOPK + j] = acc;
    }
}

__global__ void k_softmax10(const float* __restrict__ in, float* __restrict__ out) {
    int n = blockIdx.x * blockDim.x + threadIdx.x;
    if (n >= NS) return;
    float v[TOPK], m = -1e30f;
#pragma unroll
    for (int j = 0; j < TOPK; ++j) { v[j] = in[n * TOPK + j]; m = fmaxf(m, v[j]); }
    float s = 0.0f;
#pragma unroll
    for (int j = 0; j < TOPK; ++j) { v[j] = __expf(v[j] - m); s += v[j]; }
    float inv = 1.0f / s;
#pragma unroll
    for (int j = 0; j < TOPK; ++j) out[n * TOPK + j] = v[j] * inv;
}

// r_s = normal(fold_in(key,step), (NS,32)) — exact threefry counter mapping
__global__ void k_rand(float* __restrict__ r_s, unsigned k0, unsigned k1) {
    int i = blockIdx.x * blockDim.x + threadIdx.x;
    const int half = NS * RDIM / 2;
    if (i >= half) return;
    unsigned o0, o1;
    threefry2x32(k0, k1, (unsigned)i, (unsigned)(half + i), o0, o1);
    r_s[i] = bits_to_normal(o0);
    r_s[half + i] = bits_to_normal(o1);
}

// r_t[S_idx[n][j]] += S[n][j] * r_s[n]  (32 channels, 4 per thread)
__global__ void k_scatter_rt(const float* __restrict__ S, const float* __restrict__ r_s,
                             const int* __restrict__ S_idx, float* __restrict__ r_t) {
    int t = blockIdx.x * blockDim.x + threadIdx.x;
    int c4 = t & 7, nj = t >> 3;
    if (nj >= NS * TOPK) return;
    int n = nj / TOPK;
    float s = S[nj];
    int dst = S_idx[nj];
    const float4 rv = *(const float4*)(r_s + n * RDIM + c4 * 4);
    float* out = r_t + (size_t)dst * RDIM + c4 * 4;
    atomicAdd(out + 0, rv.x * s);
    atomicAdd(out + 1, rv.y * s);
    atomicAdd(out + 2, rv.z * s);
    atomicAdd(out + 3, rv.w * s);
}

// edge scatter, 32 channels
__global__ void k_scatter32(const float* __restrict__ r, const int* __restrict__ ei,
                            const float* __restrict__ ea, float* __restrict__ msg) {
    int t = blockIdx.x * blockDim.x + threadIdx.x;
    int e = t >> 3, c4 = t & 7;
    if (e >= EDG) return;
    int row = ei[e], col = ei[EDG + e];
    float w = ea[e];
    const float4 xv = *(const float4*)(r + (size_t)row * RDIM + c4 * 4);
    float* m = msg + (size_t)col * RDIM + c4 * 4;
    atomicAdd(m + 0, xv.x * w);
    atomicAdd(m + 1, xv.y * w);
    atomicAdd(m + 2, xv.z * w);
    atomicAdd(m + 3, xv.w * w);
}

// o = relu((r+msg) @ W(32x32) + b)
__global__ __launch_bounds__(256) void k_gnn2(const float* __restrict__ r, const float* __restrict__ msg,
                                              const float* __restrict__ W, const float* __restrict__ b,
                                              float* __restrict__ o) {
    __shared__ float sW[RDIM * RDIM];
    __shared__ float srm[8][RDIM];
    int tid = threadIdx.x;
    for (int i = tid; i < RDIM * RDIM; i += 256) sW[i] = W[i];
    int ln = tid >> 5, lane = tid & 31;
    int n = blockIdx.x * 8 + ln;
    srm[ln][lane] = (n < NS) ? (r[n * RDIM + lane] + msg[n * RDIM + lane]) : 0.0f;
    __syncthreads();
    if (n < NS) {
        float acc = b[lane];
#pragma unroll
        for (int c = 0; c < RDIM; ++c) acc += srm[ln][c] * sW[c * RDIM + lane];
        o[n * RDIM + lane] = fmaxf(acc, 0.0f);
    }
}

// S_hat[n][j] += mlp(o_s[n] - o_t[S_idx[n][j]])
__global__ __launch_bounds__(256) void k_mlp(const float* __restrict__ o_s, const float* __restrict__ o_t,
                                             const int* __restrict__ S_idx,
                                             const float* __restrict__ W1, const float* __restrict__ b1,
                                             const float* __restrict__ W2, const float* __restrict__ b2,
                                             float* __restrict__ S_hat) {
    __shared__ float sW1[RDIM * RDIM];
    __shared__ float sb1[RDIM];
    __shared__ float sW2[RDIM];
    int tid = threadIdx.x;
    for (int i = tid; i < RDIM * RDIM; i += 256) sW1[i] = W1[i];
    if (tid < RDIM) { sb1[tid] = b1[tid]; sW2[tid] = W2[tid]; }
    __syncthreads();
    int nj = blockIdx.x * 256 + tid;
    if (nj >= NS * TOPK) return;
    int n = nj / TOPK;
    int t = S_idx[nj];
    float D[RDIM];
#pragma unroll
    for (int c = 0; c < RDIM; ++c) D[c] = o_s[n * RDIM + c] - o_t[(size_t)t * RDIM + c];
    float out = b2[0];
#pragma unroll
    for (int cc = 0; cc < RDIM; ++cc) {
        float acc = sb1[cc];
#pragma unroll
        for (int c = 0; c < RDIM; ++c) acc += D[c] * sW1[c * RDIM + cc];
        out += fmaxf(acc, 0.0f) * sW2[cc];
    }
    S_hat[nj] += out;
}

// ================= host launcher =================
extern "C" void kernel_launch(void* const* d_in, const int* in_sizes, int n_in,
                              void* d_out, int out_size, void* d_ws, size_t ws_size,
                              hipStream_t stream) {
    (void)in_sizes; (void)n_in; (void)out_size; (void)ws_size;

    const float* x_s    = (const float*)d_in[0];
    const int*   ei_s   = (const int*)  d_in[1];
    const float* ea_s   = (const float*)d_in[2];
    const float* x_t    = (const float*)d_in[3];
    const int*   ei_t   = (const int*)  d_in[4];
    const float* ea_t   = (const float*)d_in[5];
    const float* psi1_W = (const float*)d_in[6];
    const float* psi1_b = (const float*)d_in[7];
    const float* psi2_W = (const float*)d_in[8];
    const float* psi2_b = (const float*)d_in[9];
    const float* mlp_W1 = (const float*)d_in[10];
    const float* mlp_b1 = (const float*)d_in[11];
    const float* mlp_W2 = (const float*)d_in[12];
    const float* mlp_b2 = (const float*)d_in[13];

    char* ws = (char*)d_ws;
    size_t off = 0;
    auto alloc = [&](size_t bytes) -> char* {
        char* p = ws + off;
        off = (off + bytes + 255) & ~(size_t)255;
        return p;
    };
    float*          msgbuf = (float*)alloc((size_t)NS * C_IN * 4);
    float*          h_s    = (float*)alloc((size_t)NS * C_OUT * 4);
    float*          h_t    = (float*)alloc((size_t)NT * C_OUT * 4);
    unsigned short* hs_bf  = (unsigned short*)alloc((size_t)NS * C_OUT * 2);
    unsigned short* ht_bf  = (unsigned short*)alloc((size_t)NT * C_OUT * 2);
    unsigned short* Wp     = (unsigned short*)alloc(4 * 16 * 32 * 16 * 2);
    int*            S_idx  = (int*)alloc((size_t)NS * TOPK * 4);
    float*          S_hat  = (float*)alloc((size_t)NS * TOPK * 4);
    float*          Sbuf   = (float*)alloc((size_t)NS * TOPK * 4);
    float*          r_s    = (float*)alloc((size_t)NS * RDIM * 4);
    float*          r_t    = (float*)alloc((size_t)NT * RDIM * 4);
    float*          msg2   = (float*)alloc((size_t)NS * RDIM * 4);
    float*          o_s    = (float*)alloc((size_t)NS * RDIM * 4);
    float*          o_t    = (float*)alloc((size_t)NT * RDIM * 4);
    float*          out_S0 = (float*)d_out;
    float*          out_SL = (float*)d_out + (size_t)NS * TOPK;

    // --- pack psi1 weights once per call ---
    k_packW<<<(4 * 16 * 32 * 16 + 255) / 256, 256, 0, stream>>>(psi1_W, Wp);

    // --- GNN psi1, source graph ---
    hipMemsetAsync(msgbuf, 0, (size_t)NS * C_IN * 4, stream);
    k_scatter128<<<(EDG * 32 + 255) / 256, 256, 0, stream>>>(x_s, ei_s, ea_s, msgbuf);
    k_gnn1_gemm<<<dim3(NS / 16, 4), 128, 0, stream>>>(x_s, msgbuf, Wp, psi1_b, h_s, hs_bf);

    // --- GNN psi1, target graph (reuse msg buffer) ---
    hipMemsetAsync(msgbuf, 0, (size_t)NT * C_IN * 4, stream);
    k_scatter128<<<(EDG * 32 + 255) / 256, 256, 0, stream>>>(x_t, ei_t, ea_t, msgbuf);
    k_gnn1_gemm<<<dim3(NT / 16, 4), 128, 0, stream>>>(x_t, msgbuf, Wp, psi1_b, h_t, ht_bf);

    // --- fused WMMA similarity + top-10 ---
    k_sim_topk<<<NS / 16, 128, 0, stream>>>(hs_bf, ht_bf, S_idx);

    // --- exact f32 sparse scores + S_0 ---
    k_shat<<<NS / 8, 256, 0, stream>>>(h_s, h_t, S_idx, S_hat);
    k_softmax10<<<(NS + 255) / 256, 256, 0, stream>>>(S_hat, out_S0);

    // --- refinement steps ---
    unsigned key0 = 0u, key1 = 42u;   // jax.random.key(42)
    for (int step = 0; step < 2; ++step) {
        k_softmax10<<<(NS + 255) / 256, 256, 0, stream>>>(S_hat, Sbuf);

        unsigned fk0, fk1;
        threefry2x32(key0, key1, 0u, (unsigned)step, fk0, fk1);   // fold_in on host
        k_rand<<<(NS * RDIM / 2 + 255) / 256, 256, 0, stream>>>(r_s, fk0, fk1);

        hipMemsetAsync(r_t, 0, (size_t)NT * RDIM * 4, stream);
        k_scatter_rt<<<(NS * TOPK * 8 + 255) / 256, 256, 0, stream>>>(Sbuf, r_s, S_idx, r_t);

        hipMemsetAsync(msg2, 0, (size_t)NS * RDIM * 4, stream);
        k_scatter32<<<(EDG * 8 + 255) / 256, 256, 0, stream>>>(r_s, ei_s, ea_s, msg2);
        k_gnn2<<<NS / 8, 256, 0, stream>>>(r_s, msg2, psi2_W, psi2_b, o_s);

        hipMemsetAsync(msg2, 0, (size_t)NT * RDIM * 4, stream);
        k_scatter32<<<(EDG * 8 + 255) / 256, 256, 0, stream>>>(r_t, ei_t, ea_t, msg2);
        k_gnn2<<<NT / 8, 256, 0, stream>>>(r_t, msg2, psi2_W, psi2_b, o_t);

        k_mlp<<<(NS * TOPK + 255) / 256, 256, 0, stream>>>(o_s, o_t, S_idx,
                                                           mlp_W1, mlp_b1, mlp_W2, mlp_b2, S_hat);
    }

    k_softmax10<<<(NS + 255) / 256, 256, 0, stream>>>(S_hat, out_SL);
}
